// ResnetBlockME_39307540693474
// MI455X (gfx1250) — compile-verified, matrix operand order checked by reading, not run on hardware
//
#include <hip/hip_runtime.h>
#include <math.h>

typedef __bf16 bfloat;
typedef bfloat bf16x8 __attribute__((ext_vector_type(8)));
typedef bfloat v16bf  __attribute__((ext_vector_type(16)));
typedef float  v8f    __attribute__((ext_vector_type(8)));

#define CIN1 256
#define DOUT 256
#define DIMX 128
#define KOFF 27

// ---------------------------------------------------------------------------
// prep: h0 = concat(x, time_emb[batch_ids]) in bf16
__global__ void prep_h_kernel(const float* __restrict__ x,
                              const float* __restrict__ te,
                              const int* __restrict__ bids,
                              bfloat* __restrict__ h0, int Nrows) {
    size_t total = (size_t)Nrows * 256;
    size_t i = (size_t)blockIdx.x * blockDim.x + threadIdx.x;
    size_t stride = (size_t)gridDim.x * blockDim.x;
    for (; i < total; i += stride) {
        int r = (int)(i >> 8);
        int c = (int)(i & 255);
        float v = (c < 128) ? x[(size_t)r * 128 + c]
                            : te[(size_t)bids[r] * 128 + (c - 128)];
        h0[i] = (bfloat)v;
    }
}

__global__ void prep_x_kernel(const float* __restrict__ x,
                              bfloat* __restrict__ xb, int Nrows) {
    size_t total = (size_t)Nrows * 128;
    size_t i = (size_t)blockIdx.x * blockDim.x + threadIdx.x;
    size_t stride = (size_t)gridDim.x * blockDim.x;
    for (; i < total; i += stride) xb[i] = (bfloat)x[i];
}

// W: [K][cin=256][cout=256] fp32  ->  Wt: [K][cout][cin] bf16
__global__ void transpose_w_kernel(const float* __restrict__ W,
                                   bfloat* __restrict__ Wt, int K) {
    size_t total = (size_t)K * 256 * 256;
    size_t i = (size_t)blockIdx.x * blockDim.x + threadIdx.x;
    size_t stride = (size_t)gridDim.x * blockDim.x;
    for (; i < total; i += stride) {
        size_t k = i >> 16;
        int o = (int)((i >> 8) & 255);
        int c = (int)(i & 255);
        Wt[i] = (bfloat)W[(k << 16) + ((size_t)c << 8) + o];
    }
}

// Wr: [128][256] fp32 -> Wrt: [256][128] bf16
__global__ void transpose_wr_kernel(const float* __restrict__ Wr,
                                    bfloat* __restrict__ Wrt) {
    size_t total = 256 * 128;
    size_t i = (size_t)blockIdx.x * blockDim.x + threadIdx.x;
    size_t stride = (size_t)gridDim.x * blockDim.x;
    for (; i < total; i += stride) {
        int o = (int)(i >> 7);
        int c = (int)(i & 127);
        Wrt[i] = (bfloat)Wr[(size_t)c * 256 + o];
    }
}

// ---------------------------------------------------------------------------
// Sparse conv: gather 16 rows of bf16 feat, WMMA against Wt[k], atomic-scatter
// fp32 into acc. Block = 128 threads = 4 waves; each wave owns 4 n-tiles.
__global__ __launch_bounds__(128)
void sparse_conv_kernel(const bfloat* __restrict__ feat,   // [N][256]
                        const bfloat* __restrict__ wt,     // [27][256 cout][256 cin]
                        const int* __restrict__ in_map,    // [27][M]
                        const int* __restrict__ out_map,   // [27][M]
                        float* __restrict__ acc,           // [N+1][256]
                        int M, int Nrows) {
    const int k = blockIdx.y;
    const int tileBase = blockIdx.x * 16;
    const int* im = in_map  + (size_t)k * M;
    const int* om = out_map + (size_t)k * M;

    // padding is contiguous at the map tail -> fully-padded tile: skip whole
    // block (uniform branch, before any WMMA so EXEC stays all-1s)
    if (om[tileBase] >= Nrows) return;

    const int tid  = threadIdx.x;
    const int lane = tid & 31;
    const int wave = tid >> 5;
    const int lr   = lane & 15;   // A row-in-tile / B col-in-ntile
    const int hi   = lane >> 4;   // half-wave select (K-chunk phase)
    const int ntBase = wave * 4;

    // gathered A row for this lane
    int ia = tileBase + lr;
    int gi = (ia < M) ? im[ia] : 0;
    const bfloat* arow = feat + (size_t)gi * 256;

    // scatter rows for the 8 accumulator rows this lane owns
    int orow[8];
#pragma unroll
    for (int j = 0; j < 8; ++j) {
        int idx = tileBase + j + hi * 8;
        orow[j] = (idx < M) ? om[idx] : Nrows;
    }

    const bfloat* wk = wt + (size_t)k * 256 * 256;

    v8f accv[4];
#pragma unroll
    for (int nt = 0; nt < 4; ++nt) accv[nt] = (v8f){0,0,0,0,0,0,0,0};

#pragma unroll
    for (int kt = 0; kt < 8; ++kt) {            // 256 cin / 32
        const int off1 = kt * 32 + hi * 8;
        const int off2 = off1 + 16;
        bf16x8 aLo = *(const bf16x8*)(arow + off1);
        bf16x8 aHi = *(const bf16x8*)(arow + off2);
        v16bf a = __builtin_shufflevector(aLo, aHi,
            0,1,2,3,4,5,6,7,8,9,10,11,12,13,14,15);
#pragma unroll
        for (int nt = 0; nt < 4; ++nt) {
            const bfloat* brow = wk + (size_t)((ntBase + nt) * 16 + lr) * 256;
            bf16x8 bLo = *(const bf16x8*)(brow + off1);
            bf16x8 bHi = *(const bf16x8*)(brow + off2);
            v16bf b = __builtin_shufflevector(bLo, bHi,
                0,1,2,3,4,5,6,7,8,9,10,11,12,13,14,15);
            accv[nt] = __builtin_amdgcn_wmma_f32_16x16x32_bf16(
                false, a, false, b, (short)0, accv[nt], false, false);
        }
    }

    // atomic scatter-add (skip pad rows -> avoids dump-row contention)
#pragma unroll
    for (int nt = 0; nt < 4; ++nt) {
        int col = (ntBase + nt) * 16 + lr;
#pragma unroll
        for (int j = 0; j < 8; ++j) {
            if (orow[j] < Nrows)
                unsafeAtomicAdd(acc + (size_t)orow[j] * 256 + col, accv[nt][j]);
        }
    }
}

// ---------------------------------------------------------------------------
// BN pass 1: per-channel partial sums / sumsq
__global__ __launch_bounds__(256)
void bn_reduce_kernel(const float* __restrict__ acc,
                      float* __restrict__ part, int Nrows) {
    int c = threadIdx.x;     // 256 channels
    int g = blockIdx.x;
    float s = 0.f, s2 = 0.f;
    for (int r = g; r < Nrows; r += gridDim.x) {
        float v = acc[(size_t)r * 256 + c];
        s += v; s2 += v * v;
    }
    part[(size_t)g * 512 + c]       = s;
    part[(size_t)g * 512 + 256 + c] = s2;
}

// BN pass 2: fold into per-channel scale/shift (conv bias cancels through BN)
__global__ __launch_bounds__(256)
void bn_finalize_kernel(const float* __restrict__ part,
                        const float* __restrict__ gamma,
                        const float* __restrict__ beta,
                        float* __restrict__ sc, float* __restrict__ sh,
                        int Nrows, int ngroups) {
    int c = threadIdx.x;
    float s = 0.f, s2 = 0.f;
    for (int i = 0; i < ngroups; ++i) {
        s  += part[(size_t)i * 512 + c];
        s2 += part[(size_t)i * 512 + 256 + c];
    }
    float inv = 1.0f / (float)Nrows;
    float mu  = s * inv;
    float var = s2 * inv - mu * mu;
    float rs  = rsqrtf(var + 1e-5f);
    float scale = gamma[c] * rs;
    sc[c] = scale;
    sh[c] = beta[c] - mu * scale;
}

// BN apply + SiLU -> bf16 features for next conv
__global__ void bn_apply_kernel(const float* __restrict__ acc,
                                const float* __restrict__ sc,
                                const float* __restrict__ sh,
                                bfloat* __restrict__ h, int Nrows) {
    size_t total = (size_t)Nrows * 256;
    size_t i = (size_t)blockIdx.x * blockDim.x + threadIdx.x;
    size_t stride = (size_t)gridDim.x * blockDim.x;
    for (; i < total; i += stride) {
        int c = (int)(i & 255);
        float y = acc[i] * sc[c] + sh[c];
        float v = y / (1.0f + __expf(-y));
        h[i] = (bfloat)v;
    }
}

// ---------------------------------------------------------------------------
// final: out = silu(bn2(acc)) + x @ Wr + br   (residual GEMM via WMMA)
__global__ __launch_bounds__(128)
void final_kernel(const bfloat* __restrict__ xb,    // [N][128]
                  const bfloat* __restrict__ wrt,   // [256][128]
                  const float*  __restrict__ acc,   // [N+1][256]
                  const float*  __restrict__ sc,
                  const float*  __restrict__ sh,
                  const float*  __restrict__ br,
                  float* __restrict__ out, int Nrows) {
    const int tileBase = blockIdx.x * 16;
    const int tid  = threadIdx.x;
    const int lane = tid & 31;
    const int wave = tid >> 5;
    const int lr   = lane & 15;
    const int hi   = lane >> 4;
    const int ntBase = wave * 4;

    const bfloat* arow = xb + (size_t)(tileBase + lr) * 128;

    v8f accv[4];
#pragma unroll
    for (int nt = 0; nt < 4; ++nt) accv[nt] = (v8f){0,0,0,0,0,0,0,0};

#pragma unroll
    for (int kt = 0; kt < 4; ++kt) {            // 128 cin / 32
        const int off1 = kt * 32 + hi * 8;
        const int off2 = off1 + 16;
        bf16x8 aLo = *(const bf16x8*)(arow + off1);
        bf16x8 aHi = *(const bf16x8*)(arow + off2);
        v16bf a = __builtin_shufflevector(aLo, aHi,
            0,1,2,3,4,5,6,7,8,9,10,11,12,13,14,15);
#pragma unroll
        for (int nt = 0; nt < 4; ++nt) {
            const bfloat* brow = wrt + (size_t)((ntBase + nt) * 16 + lr) * 128;
            bf16x8 bLo = *(const bf16x8*)(brow + off1);
            bf16x8 bHi = *(const bf16x8*)(brow + off2);
            v16bf b = __builtin_shufflevector(bLo, bHi,
                0,1,2,3,4,5,6,7,8,9,10,11,12,13,14,15);
            accv[nt] = __builtin_amdgcn_wmma_f32_16x16x32_bf16(
                false, a, false, b, (short)0, accv[nt], false, false);
        }
    }

#pragma unroll
    for (int nt = 0; nt < 4; ++nt) {
        int col = (ntBase + nt) * 16 + lr;
#pragma unroll
        for (int j = 0; j < 8; ++j) {
            int row = tileBase + j + hi * 8;
            size_t idx = (size_t)row * 256 + col;
            float y = acc[idx] * sc[col] + sh[col];
            float s = y / (1.0f + __expf(-y));
            out[idx] = s + accv[nt][j] + br[col];
        }
    }
}

// ---------------------------------------------------------------------------
static inline size_t align256(size_t x) { return (x + 255) & ~(size_t)255; }

extern "C" void kernel_launch(void* const* d_in, const int* in_sizes, int n_in,
                              void* d_out, int out_size, void* d_ws, size_t ws_size,
                              hipStream_t stream) {
    const float* x    = (const float*)d_in[0];
    const float* te   = (const float*)d_in[1];
    const int*   bids = (const int*)d_in[2];
    const int*   imap = (const int*)d_in[3];
    const int*   omap = (const int*)d_in[4];
    const float* W1   = (const float*)d_in[5];
    const float* g1   = (const float*)d_in[7];
    const float* be1  = (const float*)d_in[8];
    const float* W2   = (const float*)d_in[9];
    const float* g2   = (const float*)d_in[11];
    const float* be2  = (const float*)d_in[12];
    const float* Wr   = (const float*)d_in[13];
    const float* br   = (const float*)d_in[14];
    float* out = (float*)d_out;

    const int Nrows = in_sizes[0] / 128;   // 262144
    const int M     = in_sizes[3] / 27;    // padded map width
    const int NG    = 1024;                // BN reduction groups

    // workspace layout
    char* ws = (char*)d_ws;
    size_t o = 0;
    bfloat* hbuf = (bfloat*)(ws + o); o = align256(o + (size_t)Nrows * 256 * 2);
    bfloat* xb   = (bfloat*)(ws + o); o = align256(o + (size_t)Nrows * 128 * 2);
    float*  acc  = (float*) (ws + o); o = align256(o + ((size_t)Nrows + 1) * 256 * 4);
    bfloat* wt1  = (bfloat*)(ws + o); o = align256(o + (size_t)KOFF * 256 * 256 * 2);
    bfloat* wt2  = (bfloat*)(ws + o); o = align256(o + (size_t)KOFF * 256 * 256 * 2);
    bfloat* wrt  = (bfloat*)(ws + o); o = align256(o + (size_t)256 * 128 * 2);
    float*  part = (float*) (ws + o); o = align256(o + (size_t)NG * 512 * 4);
    float*  sc   = (float*) (ws + o); o = align256(o + 256 * 4);
    float*  sh   = (float*) (ws + o); o = align256(o + 256 * 4);
    (void)ws_size; (void)n_in; (void)out_size;

    const size_t accBytes = ((size_t)Nrows + 1) * 256 * 4;
    const int tilesM = (M + 15) / 16;

    // ---- prep: bf16 activations + transposed bf16 weights
    prep_h_kernel<<<4096, 256, 0, stream>>>(x, te, bids, hbuf, Nrows);
    prep_x_kernel<<<2048, 256, 0, stream>>>(x, xb, Nrows);
    transpose_w_kernel<<<2048, 256, 0, stream>>>(W1, wt1, KOFF);
    transpose_w_kernel<<<2048, 256, 0, stream>>>(W2, wt2, KOFF);
    transpose_wr_kernel<<<128, 256, 0, stream>>>(Wr, wrt);

    // ---- block 1: sparse conv -> BN -> SiLU
    hipMemsetAsync(acc, 0, accBytes, stream);
    sparse_conv_kernel<<<dim3(tilesM, KOFF), 128, 0, stream>>>(
        hbuf, wt1, imap, omap, acc, M, Nrows);
    bn_reduce_kernel<<<NG, 256, 0, stream>>>(acc, part, Nrows);
    bn_finalize_kernel<<<1, 256, 0, stream>>>(part, g1, be1, sc, sh, Nrows, NG);
    bn_apply_kernel<<<4096, 256, 0, stream>>>(acc, sc, sh, hbuf, Nrows);

    // ---- block 2: sparse conv -> BN stats
    hipMemsetAsync(acc, 0, accBytes, stream);
    sparse_conv_kernel<<<dim3(tilesM, KOFF), 128, 0, stream>>>(
        hbuf, wt2, imap, omap, acc, M, Nrows);
    bn_reduce_kernel<<<NG, 256, 0, stream>>>(acc, part, Nrows);
    bn_finalize_kernel<<<1, 256, 0, stream>>>(part, g2, be2, sc, sh, Nrows, NG);

    // ---- fused BN2+SiLU + residual WMMA GEMM + bias
    final_kernel<<<Nrows / 16, 128, 0, stream>>>(
        xb, wrt, acc, sc, sh, br, out, Nrows);
}
